// DSVDD_9972914061970
// MI455X (gfx1250) — matrix operand order repaired and checked by reading, not compiled, not run on hardware
//
#include <hip/hip_runtime.h>
#include <hip/hip_bf16.h>

typedef __bf16 bf16_t;
typedef __attribute__((ext_vector_type(16))) __bf16 v16bf;
typedef __attribute__((ext_vector_type(8)))  __bf16 v8bf;
typedef __attribute__((ext_vector_type(8)))  float  v8f;

#define B_      8
#define H_      56
#define HW_     3136      // 56*56 pixels
#define KREAL_  1794      // 256 + 512 + 1024 + 2 coord channels
#define KPAD_   1824      // padded to multiple of 32 for WMMA K-loop
#define O_      448       // conv output channels == center feature dim
#define KS1_    57        // KPAD_/32
#define KS2_    14        // O_/32
#define MT_     196       // HW_/16  (M tiles)
#define NT1_    28        // O_/16   (N tiles, gemm1)
#define NT2_    196       // HW_/16  (N tiles = center tiles, gemm2)

static __device__ __forceinline__ int iclamp(int v, int lo, int hi) {
    return v < lo ? lo : (v > hi ? hi : v);
}

// 3x3 avg pool, stride 1, zero pad, count_include_pad -> sum/9
static __device__ __forceinline__ float pool3(const float* __restrict__ plane, int S, int y, int x) {
    float s = 0.f;
    for (int dy = -1; dy <= 1; ++dy) {
        int yy = y + dy;
        if (yy < 0 || yy >= S) continue;
        for (int dx = -1; dx <= 1; ++dx) {
            int xx = x + dx;
            if (xx < 0 || xx >= S) continue;
            s += plane[yy * S + xx];
        }
    }
    return s * (1.f / 9.f);
}

// half-pixel bilinear upsample of pooled SxS plane, sampled at output (h,w) of 56x56
static __device__ __forceinline__ float up_pool(const float* __restrict__ plane, int S, float scale, int h, int w) {
    float fy = ((float)h + 0.5f) * scale - 0.5f;
    float fx = ((float)w + 0.5f) * scale - 0.5f;
    int y0 = (int)floorf(fy); float wy = fy - (float)y0;
    int x0 = (int)floorf(fx); float wx = fx - (float)x0;
    int y0c = iclamp(y0,     0, S - 1);
    int y1c = iclamp(y0 + 1, 0, S - 1);
    int x0c = iclamp(x0,     0, S - 1);
    int x1c = iclamp(x0 + 1, 0, S - 1);
    float p00 = pool3(plane, S, y0c, x0c);
    float p01 = pool3(plane, S, y0c, x1c);
    float p10 = pool3(plane, S, y1c, x0c);
    float p11 = pool3(plane, S, y1c, x1c);
    return (1.f - wy) * ((1.f - wx) * p00 + wx * p01) + wy * ((1.f - wx) * p10 + wx * p11);
}

// Build X[b, n(pix), k] in bf16, K-contiguous (row-major), padded to KPAD_ with zeros.
__global__ void prep_x_kernel(const float* __restrict__ p1, const float* __restrict__ p2,
                              const float* __restrict__ p3, bf16_t* __restrict__ Xbf) {
    long long idx = (long long)blockIdx.x * blockDim.x + threadIdx.x;
    const long long total = (long long)B_ * HW_ * KPAD_;
    if (idx >= total) return;
    int k = (int)(idx % KPAD_);
    long long r = idx / KPAD_;
    int n = (int)(r % HW_);
    int b = (int)(r / HW_);
    int h = n / H_, w = n % H_;
    float val;
    if (k < 256) {
        const float* plane = p1 + (long long)(b * 256 + k) * HW_;
        val = pool3(plane, 56, h, w);
    } else if (k < 768) {
        const float* plane = p2 + (long long)(b * 512 + (k - 256)) * (28 * 28);
        val = up_pool(plane, 28, 0.5f, h, w);
    } else if (k < 1792) {
        const float* plane = p3 + (long long)(b * 1024 + (k - 768)) * (14 * 14);
        val = up_pool(plane, 14, 0.25f, h, w);
    } else if (k == 1792) {
        val = ((float)h / 55.f) * 2.f - 1.f;   // xx varies along rows
    } else if (k == 1793) {
        val = ((float)w / 55.f) * 2.f - 1.f;   // yy varies along cols
    } else {
        val = 0.f;                              // K padding
    }
    Xbf[idx] = (bf16_t)val;
}

// W[448,1794] f32 -> Wbf[448,1824] bf16 (zero padded K)
__global__ void prep_w_kernel(const float* __restrict__ W, bf16_t* __restrict__ Wbf) {
    int idx = blockIdx.x * blockDim.x + threadIdx.x;
    if (idx >= O_ * KPAD_) return;
    int o = idx / KPAD_, k = idx % KPAD_;
    float v = (k < KREAL_) ? W[(long long)o * KREAL_ + k] : 0.f;
    Wbf[idx] = (bf16_t)v;
}

// C[448,3136] f32 -> Ct[3136,448] bf16 (K=448 contiguous per center)
__global__ void prep_ct_kernel(const float* __restrict__ C, bf16_t* __restrict__ Ct) {
    int idx = blockIdx.x * blockDim.x + threadIdx.x;
    if (idx >= HW_ * O_) return;
    int m = idx / O_, o = idx % O_;
    Ct[idx] = (bf16_t)C[(long long)o * HW_ + m];
}

// cnorm[m] = sum_o C[o,m]^2
__global__ void cnorm_kernel(const float* __restrict__ C, float* __restrict__ cnorm) {
    int m = blockIdx.x * blockDim.x + threadIdx.x;
    if (m >= HW_) return;
    float s = 0.f;
    for (int o = 0; o < O_; ++o) {
        float v = C[(long long)o * HW_ + m];
        s += v * v;
    }
    cnorm[m] = s;
}

// GEMM1: phi[b, n, o] = sum_k X[b,n,k] * W[o,k] + bias[o]   (NT, bf16 WMMA, f32 acc)
__global__ void __launch_bounds__(256) gemm1_kernel(const bf16_t* __restrict__ X,
                                                    const bf16_t* __restrict__ Wb,
                                                    const float* __restrict__ bias,
                                                    bf16_t* __restrict__ phi) {
    int wave = (blockIdx.x << 3) + (threadIdx.x >> 5);   // 8 waves/block, one 16x16 tile each
    int lane = threadIdx.x & 31;
    int b  = wave / (MT_ * NT1_);
    int t  = wave % (MT_ * NT1_);
    int tm = t / NT1_, tn = t % NT1_;
    int lhalf = lane >> 4;
    int l15   = lane & 15;
    // A fragment: lane = M row, elems 0..7 at K = 32*ks + 8*lhalf, elems 8..15 at +16
    const bf16_t* Arow = X + ((long long)b * HW_ + tm * 16 + l15) * KPAD_ + (lhalf << 3);
    // B fragment: lane = N col, elems 0..7 at K = 32*ks + 8*lhalf, elems 8..15 at +16
    const bf16_t* Brow = Wb + (long long)(tn * 16 + l15) * KPAD_ + (lhalf << 3);
    v8f acc = {};
    for (int ks = 0; ks < KS1_; ++ks) {
        int k0 = ks * 32;
        v8bf alo = *(const v8bf*)(Arow + k0);
        v8bf ahi = *(const v8bf*)(Arow + k0 + 16);
        v8bf blo = *(const v8bf*)(Brow + k0);
        v8bf bhi = *(const v8bf*)(Brow + k0 + 16);
        v16bf a  = __builtin_shufflevector(alo, ahi, 0,1,2,3,4,5,6,7,8,9,10,11,12,13,14,15);
        v16bf bb = __builtin_shufflevector(blo, bhi, 0,1,2,3,4,5,6,7,8,9,10,11,12,13,14,15);
        acc = __builtin_amdgcn_wmma_f32_16x16x32_bf16(false, a, false, bb, (short)0, acc, false, false);
    }
    float bv = bias[tn * 16 + l15];
    int rbase = tm * 16 + (lhalf << 3);   // D: VGPR g -> M row = 8*lhalf + g, N = l15
    #pragma unroll
    for (int g = 0; g < 8; ++g) {
        float v = acc[g] + bv;
        phi[((long long)b * HW_ + rbase + g) * O_ + tn * 16 + l15] = (bf16_t)v;
    }
}

// feat[row] = sum_o phi[row,o]^2  (one wave per row)
__global__ void feat_kernel(const bf16_t* __restrict__ phi, float* __restrict__ feat) {
    int wid  = (blockIdx.x * blockDim.x + threadIdx.x) >> 5;   // 25088 rows total
    int lane = threadIdx.x & 31;
    const bf16_t* row = phi + (long long)wid * O_;
    float s = 0.f;
    for (int j = lane; j < O_; j += 32) {
        float v = (float)row[j];
        s += v * v;
    }
    #pragma unroll
    for (int m = 16; m > 0; m >>= 1) s += __shfl_xor(s, m, 32);
    if (lane == 0) feat[wid] = s;
}

// GEMM2 + fused sqrt-distance + running top-3 + softmin score.
// Block = 4 waves; each wave owns 16 pixel rows and sweeps all 196 center tiles.
__global__ void __launch_bounds__(128) gemm2_kernel(const bf16_t* __restrict__ phi,
                                                    const bf16_t* __restrict__ Ct,
                                                    const float* __restrict__ feat,
                                                    const float* __restrict__ cnorm,
                                                    float* __restrict__ out) {
    __shared__ float sm[4][16][16][3];
    int wv   = threadIdx.x >> 5;
    int lane = threadIdx.x & 31;
    int b  = blockIdx.x / 49;
    int rb = blockIdx.x % 49;
    int rowBase = rb * 64 + wv * 16;
    int lhalf = lane >> 4, l15 = lane & 15;

    // Preload all A fragments for these 16 rows (K = 448 -> 14 fragments)
    const bf16_t* Arow = phi + ((long long)b * HW_ + rowBase + l15) * O_ + (lhalf << 3);
    v16bf afrag[KS2_];
    #pragma unroll
    for (int ks = 0; ks < KS2_; ++ks) {
        v8bf lo = *(const v8bf*)(Arow + ks * 32);
        v8bf hi = *(const v8bf*)(Arow + ks * 32 + 16);
        afrag[ks] = __builtin_shufflevector(lo, hi, 0,1,2,3,4,5,6,7,8,9,10,11,12,13,14,15);
    }

    float frow[8], t0[8], t1[8], t2[8];
    #pragma unroll
    for (int g = 0; g < 8; ++g) {
        frow[g] = feat[(long long)b * HW_ + rowBase + (lhalf << 3) + g];
        t0[g] = t1[g] = t2[g] = 3.4e38f;
    }

    for (int tn = 0; tn < NT2_; ++tn) {
        const bf16_t* Brow = Ct + (long long)(tn * 16 + l15) * O_ + (lhalf << 3);
        v8f acc = {};
        #pragma unroll
        for (int ks = 0; ks < KS2_; ++ks) {
            v8bf lo = *(const v8bf*)(Brow + ks * 32);
            v8bf hi = *(const v8bf*)(Brow + ks * 32 + 16);
            v16bf bb = __builtin_shufflevector(lo, hi, 0,1,2,3,4,5,6,7,8,9,10,11,12,13,14,15);
            acc = __builtin_amdgcn_wmma_f32_16x16x32_bf16(false, afrag[ks], false, bb, (short)0, acc, false, false);
        }
        float cn = cnorm[tn * 16 + l15];   // this lane's center column
        #pragma unroll
        for (int g = 0; g < 8; ++g) {
            float d2 = frow[g] + cn - 2.f * acc[g];
            float d  = sqrtf(fmaxf(d2, 0.f));
            if (d < t0[g])      { t2[g] = t1[g]; t1[g] = t0[g]; t0[g] = d; }
            else if (d < t1[g]) { t2[g] = t1[g]; t1[g] = d; }
            else if (d < t2[g]) { t2[g] = d; }
        }
    }

    // Cross-lane merge: 16 lanes x top-3 candidates per row -> global top-3
    #pragma unroll
    for (int g = 0; g < 8; ++g) {
        int rl = (lhalf << 3) + g;
        sm[wv][rl][l15][0] = t0[g];
        sm[wv][rl][l15][1] = t1[g];
        sm[wv][rl][l15][2] = t2[g];
    }
    __syncthreads();
    if (lane < 16) {
        float a0 = 3.4e38f, a1 = 3.4e38f, a2 = 3.4e38f;
        for (int c = 0; c < 16; ++c) {
            #pragma unroll
            for (int j = 0; j < 3; ++j) {
                float d = sm[wv][lane][c][j];
                if (d < a0)      { a2 = a1; a1 = a0; a0 = d; }
                else if (d < a1) { a2 = a1; a1 = d; }
                else if (d < a2) { a2 = d; }
            }
        }
        // softmin weighting: exp(-a0)/(exp(-a0)+exp(-a1)+exp(-a2)) * a0
        float score = a0 / (1.f + __expf(a0 - a1) + __expf(a0 - a2));
        out[(long long)b * HW_ + rowBase + lane] = score;
    }
}

extern "C" void kernel_launch(void* const* d_in, const int* in_sizes, int n_in,
                              void* d_out, int out_size, void* d_ws, size_t ws_size,
                              hipStream_t stream) {
    const float* p1   = (const float*)d_in[0];   // [8,256,56,56]
    const float* p2   = (const float*)d_in[1];   // [8,512,28,28]
    const float* p3   = (const float*)d_in[2];   // [8,1024,14,14]
    const float* W    = (const float*)d_in[3];   // [448,1794]
    const float* bias = (const float*)d_in[4];   // [448]
    const float* C    = (const float*)d_in[5];   // [448,3136]
    float* out = (float*)d_out;                  // [8,3136]

    char* ws = (char*)d_ws;
    size_t off = 0;
    auto carve = [&](size_t bytes) -> char* {
        char* p = ws + off;
        off = (off + bytes + 255) & ~(size_t)255;
        return p;
    };
    bf16_t* Xbf  = (bf16_t*)carve((size_t)B_ * HW_ * KPAD_ * 2);  // ~91.5 MB
    bf16_t* Wbf  = (bf16_t*)carve((size_t)O_ * KPAD_ * 2);        // ~1.6 MB
    bf16_t* phi  = (bf16_t*)carve((size_t)B_ * HW_ * O_ * 2);     // ~22.5 MB
    bf16_t* Ct   = (bf16_t*)carve((size_t)HW_ * O_ * 2);          // ~2.8 MB
    float*  feat = (float*)carve((size_t)B_ * HW_ * 4);
    float*  cn   = (float*)carve((size_t)HW_ * 4);
    (void)ws_size; (void)in_sizes; (void)n_in; (void)out_size;

    {   // descriptor build (pools + bilinear + coords) -> bf16
        long long total = (long long)B_ * HW_ * KPAD_;
        int blocks = (int)((total + 255) / 256);
        prep_x_kernel<<<blocks, 256, 0, stream>>>(p1, p2, p3, Xbf);
    }
    prep_w_kernel<<<(O_ * KPAD_ + 255) / 256, 256, 0, stream>>>(W, Wbf);
    prep_ct_kernel<<<(HW_ * O_ + 255) / 256, 256, 0, stream>>>(C, Ct);
    cnorm_kernel<<<(HW_ + 255) / 256, 256, 0, stream>>>(C, cn);

    // GEMM1: 8 batches * 196 * 28 tiles = 43904 wave-tiles, 8 waves/block
    gemm1_kernel<<<(B_ * MT_ * NT1_) / 8, 256, 0, stream>>>(Xbf, Wbf, bias, phi);

    // row norms of phi: 25088 rows, one wave each
    feat_kernel<<<(B_ * HW_) / 8, 256, 0, stream>>>(phi, feat);

    // GEMM2 + fused top-3 + score: 8 * 49 blocks of 4 waves (64 rows each)
    gemm2_kernel<<<B_ * 49, 128, 0, stream>>>(phi, Ct, feat, cn, out);
}
